// SelfAttentionBlock_84215718740182
// MI455X (gfx1250) — matrix operand order repaired and checked by reading, not compiled, not run on hardware
//
#include <hip/hip_runtime.h>
#include <hip/hip_bf16.h>

typedef __attribute__((ext_vector_type(16))) __bf16          v16bf;
typedef __attribute__((ext_vector_type(16))) unsigned short  v16u;
typedef __attribute__((ext_vector_type(8)))  float           v8f;
typedef __attribute__((ext_vector_type(4)))  int             v4i;

#define BD     2
#define SEQ    2048
#define DIM    1024
#define NHEAD  16
#define DH     64
#define QKDIM  3072
#define MTOT   (BD * SEQ)   // 4096

// ---------------- async global->LDS copy (CDNA5), guarded ----------------
#if __has_builtin(__builtin_amdgcn_global_load_async_to_lds_b128)
#define ASYNC_OK 1
#else
#define ASYNC_OK 0
#endif

__device__ __forceinline__ void async_cp16(const unsigned short* g, unsigned short* l) {
#if ASYNC_OK
  __builtin_amdgcn_global_load_async_to_lds_b128(
      (__attribute__((address_space(1))) v4i*)(g),
      (__attribute__((address_space(3))) v4i*)(l), 0, 0);
#else
  *(uint4*)l = *(const uint4*)g;   // synchronous fallback
#endif
}

__device__ __forceinline__ void wait_async0() {
#if ASYNC_OK
#if __has_builtin(__builtin_amdgcn_s_wait_asynccnt)
  __builtin_amdgcn_s_wait_asynccnt(0);
#else
  asm volatile("s_wait_asynccnt 0x0" ::: "memory");
#endif
#endif
}

// ---------- bf16 helpers (round-to-nearest-even) ----------
__device__ __forceinline__ unsigned short f2bf(float f) {
  union { float f; unsigned int u; } x; x.f = f;
  unsigned int u = x.u;
  unsigned int r = u + 0x7fffu + ((u >> 16) & 1u);
  return (unsigned short)(r >> 16);
}

__global__ void cvt_f32_bf16(const float* __restrict__ src,
                             unsigned short* __restrict__ dst, int n) {
  int i = blockIdx.x * blockDim.x + threadIdx.x;
  if (i < n) dst[i] = f2bf(src[i]);
}

// ---------- fragment loaders (wave32 WMMA operands; contiguous k order shared by A and B) ----------
// A (16x32 bf16) from row-major tile: lane r=lane&15 is row; lanes<16 hold
// K 0..15, lanes>=16 hold K 16..31; VGPR j = (K,K+1) pair.
__device__ __forceinline__ v16bf load_a_frag(const unsigned short* p, int stride) {
  int lane = threadIdx.x & 31;
  int r  = lane & 15;
  int kb = (lane >> 4) << 4;
  v16u u;
#pragma unroll
  for (int j = 0; j < 8; ++j) {
    int kk = kb + 2 * j;
    u[2 * j]     = p[r * stride + kk];
    u[2 * j + 1] = p[r * stride + kk + 1];
  }
  return __builtin_bit_cast(v16bf, u);
}

// B (32x16 bf16) from ROW-MAJOR KxN tile: B[k][c] at p[k*stride + col0 + c].
// lane c=lane&15 is column; lanes<16 hold K 0..15, lanes>=16 hold K 16..31.
__device__ __forceinline__ v16bf load_b_frag_rm(const unsigned short* p, int stride,
                                                int col0, int k0) {
  int lane = threadIdx.x & 31;
  int c  = col0 + (lane & 15);
  int kb = k0 + ((lane >> 4) << 4);
  v16u u;
#pragma unroll
  for (int j = 0; j < 8; ++j) {
    int kk = kb + 2 * j;
    u[2 * j]     = p[kk * stride + c];
    u[2 * j + 1] = p[(kk + 1) * stride + c];
  }
  return __builtin_bit_cast(v16bf, u);
}

// B (32x16 bf16) where the tile is stored transposed (column-contiguous-K):
// B[k][c] at p[(col0+c)*stride + k0 + k]. Used for S = Q @ K^T (K row-major).
__device__ __forceinline__ v16bf load_b_frag_tr(const unsigned short* p, int stride,
                                                int col0, int k0) {
  int lane = threadIdx.x & 31;
  int c  = col0 + (lane & 15);
  int kb = k0 + ((lane >> 4) << 4);
  v16u u;
#pragma unroll
  for (int j = 0; j < 8; ++j) {
    int kk = kb + 2 * j;
    u[2 * j]     = p[c * stride + kk];
    u[2 * j + 1] = p[c * stride + kk + 1];
  }
  return __builtin_bit_cast(v16bf, u);
}

// ---------- tiled bf16 GEMM: C[M,Nc] = A[M,K] @ Bw[K,Nc] + bias ----------
// block tile 128x64, K-step 32, 8 waves each doing 2x2 WMMA tiles.
// Double-buffered async global->LDS staging; both tiles row-major in LDS.
template <int OUT_F32>
__global__ __launch_bounds__(256) void gemm_bf16(
    const unsigned short* __restrict__ A, const unsigned short* __restrict__ Bw,
    const float* __restrict__ bias, unsigned short* __restrict__ Cb,
    float* __restrict__ Cf, int M, int K, int Nc) {
  __shared__ __align__(16) unsigned short As[2][128 * 40];  // A[row][k], stride 40 (80B)
  __shared__ __align__(16) unsigned short Ws[2][32 * 72];   // B[k][n],  stride 72 (144B)

  const int tid  = threadIdx.x;
  const int wid  = tid >> 5;
  const int lane = tid & 31;
  const int m0 = blockIdx.y * 128;
  const int n0 = blockIdx.x * 64;
  const int wm = (wid & 3) * 32;
  const int wn = (wid >> 2) * 32;

  v8f acc[2][2];
#pragma unroll
  for (int i = 0; i < 2; ++i)
#pragma unroll
    for (int j = 0; j < 2; ++j)
      acc[i][j] = v8f{0.f, 0.f, 0.f, 0.f, 0.f, 0.f, 0.f, 0.f};

  // stage one 128x32 A tile + 32x64 B tile into buffer `buf` (16B per lane)
  auto stage = [&](int kt, int buf) {
#pragma unroll
    for (int i = 0; i < 2; ++i) {                 // 512 chunks of A
      int li  = tid + 256 * i;
      int row = li >> 2;
      int seg = li & 3;
      async_cp16(A + (size_t)(m0 + row) * K + kt + seg * 8,
                 &As[buf][row * 40 + seg * 8]);
    }
    {                                             // 256 chunks of B
      int row = tid >> 3;
      int seg = tid & 7;
      async_cp16(Bw + (size_t)(kt + row) * Nc + n0 + seg * 8,
                 &Ws[buf][row * 72 + seg * 8]);
    }
  };

  stage(0, 0);
  const int nIter = K / 32;
  for (int it = 0; it < nIter; ++it) {
    const int cur = it & 1;
    wait_async0();
    __syncthreads();
    if (it + 1 < nIter) stage((it + 1) * 32, cur ^ 1);

    v16bf a0 = load_a_frag(&As[cur][(wm + 0)  * 40], 40);
    v16bf a1 = load_a_frag(&As[cur][(wm + 16) * 40], 40);
    v16bf b0 = load_b_frag_rm(&Ws[cur][0], 72, wn + 0, 0);
    v16bf b1 = load_b_frag_rm(&Ws[cur][0], 72, wn + 16, 0);

    acc[0][0] = __builtin_amdgcn_wmma_f32_16x16x32_bf16(false, a0, false, b0, (short)0, acc[0][0], false, false);
    acc[0][1] = __builtin_amdgcn_wmma_f32_16x16x32_bf16(false, a0, false, b1, (short)0, acc[0][1], false, false);
    acc[1][0] = __builtin_amdgcn_wmma_f32_16x16x32_bf16(false, a1, false, b0, (short)0, acc[1][0], false, false);
    acc[1][1] = __builtin_amdgcn_wmma_f32_16x16x32_bf16(false, a1, false, b1, (short)0, acc[1][1], false, false);
  }

  const int cc = lane & 15;
  const int rh = (lane >> 4) * 8;
#pragma unroll
  for (int tm = 0; tm < 2; ++tm)
#pragma unroll
    for (int tn = 0; tn < 2; ++tn) {
      int col = n0 + wn + tn * 16 + cc;
      float bv = bias[col];
#pragma unroll
      for (int j = 0; j < 8; ++j) {
        int row = m0 + wm + tm * 16 + rh + j;
        float v = acc[tm][tn][j] + bv;
        if (OUT_F32) Cf[(size_t)row * Nc + col] = v;
        else         Cb[(size_t)row * Nc + col] = f2bf(v);
      }
    }
}

// ---------- flash attention: per wave 16 queries, key blocks of 32 ----------
__global__ __launch_bounds__(256) void flash_attn(
    const unsigned short* __restrict__ qkv, unsigned short* __restrict__ Ob,
    const int* __restrict__ causal_flag) {
  __shared__ __align__(16) unsigned short Ks[2][32 * 72];   // K[key][dh], stride 72
  __shared__ __align__(16) unsigned short Vs[2][32 * 72];   // V[key][dh], stride 72
  __shared__ __align__(16) unsigned short Ps[8 * 16 * 40];  // per-wave P bounce

  const int tid  = threadIdx.x;
  const int w    = tid >> 5;
  const int lane = tid & 31;
  const int bh = blockIdx.x;
  const int b  = bh >> 4;
  const int h  = bh & 15;
  const int qbase = blockIdx.y * 128;
  const int qrow0 = qbase + w * 16;
  const int causal = causal_flag[0];

  const size_t rs = QKDIM;
  const unsigned short* Qb = qkv + (size_t)(b * SEQ) * rs + h * DH;
  const unsigned short* Kb = Qb + DIM;
  const unsigned short* Vb = Qb + 2 * DIM;

  // Q fragments (A layout), dh halves 0..31 / 32..63, loaded once.
  v16bf aq[2];
  {
    int r  = lane & 15;
    int kb = (lane >> 4) << 4;
    const unsigned short* qp = Qb + (size_t)(qrow0 + r) * rs;
#pragma unroll
    for (int s = 0; s < 2; ++s) {
      v16u u;
#pragma unroll
      for (int j = 0; j < 8; ++j) {
        int kk = kb + 2 * j;
        u[2 * j]     = qp[s * 32 + kk];
        u[2 * j + 1] = qp[s * 32 + kk + 1];
      }
      aq[s] = __builtin_bit_cast(v16bf, u);
    }
  }

  float m_run[8], l_run[8];
  v8f o_acc[4];
#pragma unroll
  for (int j = 0; j < 8; ++j) { m_run[j] = -1e30f; l_run[j] = 0.f; }
#pragma unroll
  for (int t = 0; t < 4; ++t) o_acc[t] = v8f{0.f, 0.f, 0.f, 0.f, 0.f, 0.f, 0.f, 0.f};

  const float scale = 0.125f;  // 1/sqrt(64)
  const int qmax = qrow0 + 15;
  const int nkb = causal ? (qbase + 128) / 32 : (SEQ / 32);

  // stage one 32x64 K tile + V tile (16B per lane per tile)
  auto stage = [&](int kbi, int buf) {
    int kk  = tid >> 3;
    int seg = tid & 7;
    async_cp16(Kb + (size_t)(kbi * 32 + kk) * rs + seg * 8, &Ks[buf][kk * 72 + seg * 8]);
    async_cp16(Vb + (size_t)(kbi * 32 + kk) * rs + seg * 8, &Vs[buf][kk * 72 + seg * 8]);
  };

  stage(0, 0);
  for (int kbi = 0; kbi < nkb; ++kbi) {
    const int kb32 = kbi * 32;
    const int cur  = kbi & 1;
    wait_async0();
    __syncthreads();
    if (kbi + 1 < nkb) stage(kbi + 1, cur ^ 1);

    if (!causal || kb32 <= qmax) {       // wave-uniform => EXEC all-ones inside
      const int c  = lane & 15;
      const int rh = (lane >> 4) * 8;

      // S = Q @ K^T : two 16-key column tiles, 2 K-steps (dh) each
      v8f s0 = v8f{0.f,0.f,0.f,0.f,0.f,0.f,0.f,0.f};
      v8f s1 = v8f{0.f,0.f,0.f,0.f,0.f,0.f,0.f,0.f};
#pragma unroll
      for (int nt = 0; nt < 2; ++nt) {
        v8f sacc = v8f{0.f,0.f,0.f,0.f,0.f,0.f,0.f,0.f};
#pragma unroll
        for (int s = 0; s < 2; ++s) {
          v16bf bk = load_b_frag_tr(&Ks[cur][0], 72, nt * 16, s * 32);
          sacc = __builtin_amdgcn_wmma_f32_16x16x32_bf16(false, aq[s], false, bk, (short)0, sacc, false, false);
        }
        if (nt == 0) s0 = sacc; else s1 = sacc;
      }

      // scale + causal mask
#pragma unroll
      for (int j = 0; j < 8; ++j) {
        int q = qrow0 + rh + j;
        float v0 = s0[j] * scale;
        float v1 = s1[j] * scale;
        if (causal) {
          if (kb32 + c > q)      v0 = -1e30f;
          if (kb32 + 16 + c > q) v1 = -1e30f;
        }
        s0[j] = v0; s1[j] = v1;
      }

      // online softmax (row reductions across 16 lanes, wave32-safe)
      float alpha[8];
#pragma unroll
      for (int j = 0; j < 8; ++j) {
        float mx = fmaxf(s0[j], s1[j]);
        mx = fmaxf(mx, __shfl_xor(mx, 1, 32));
        mx = fmaxf(mx, __shfl_xor(mx, 2, 32));
        mx = fmaxf(mx, __shfl_xor(mx, 4, 32));
        mx = fmaxf(mx, __shfl_xor(mx, 8, 32));
        float mnew = fmaxf(m_run[j], mx);
        alpha[j] = __expf(m_run[j] - mnew);
        m_run[j] = mnew;
        float p0 = __expf(s0[j] - mnew);
        float p1 = __expf(s1[j] - mnew);
        s0[j] = p0; s1[j] = p1;
        float rsum = p0 + p1;
        rsum += __shfl_xor(rsum, 1, 32);
        rsum += __shfl_xor(rsum, 2, 32);
        rsum += __shfl_xor(rsum, 4, 32);
        rsum += __shfl_xor(rsum, 8, 32);
        l_run[j] = l_run[j] * alpha[j] + rsum;
      }
#pragma unroll
      for (int t = 0; t < 4; ++t)
#pragma unroll
        for (int j = 0; j < 8; ++j) o_acc[t][j] *= alpha[j];

      // P: C-layout -> A-layout via per-wave LDS bounce (same-wave LDS in-order)
      unsigned short* Pw = &Ps[w * 16 * 40];
#pragma unroll
      for (int j = 0; j < 8; ++j) {
        int rj = rh + j;
        Pw[rj * 40 + c]      = f2bf(s0[j]);
        Pw[rj * 40 + 16 + c] = f2bf(s1[j]);
      }
      v16bf ap = load_a_frag(Pw, 40);

      // O += P @ V : 4 dh tiles of 16 (V row-major in LDS)
#pragma unroll
      for (int vt = 0; vt < 4; ++vt) {
        v16bf bv = load_b_frag_rm(&Vs[cur][0], 72, vt * 16, 0);
        o_acc[vt] = __builtin_amdgcn_wmma_f32_16x16x32_bf16(false, ap, false, bv, (short)0, o_acc[vt], false, false);
      }
    }
  }

  // normalize + store bf16 attention output (heads re-interleaved)
  {
    const int c  = lane & 15;
    const int rh = (lane >> 4) * 8;
#pragma unroll
    for (int vt = 0; vt < 4; ++vt)
#pragma unroll
      for (int j = 0; j < 8; ++j) {
        int q = qrow0 + rh + j;
        float inv = 1.f / l_run[j];
        Ob[(size_t)(b * SEQ + q) * DIM + h * DH + vt * 16 + c] = f2bf(o_acc[vt][j] * inv);
      }
  }
}

// ---------- host-side launch ----------
extern "C" void kernel_launch(void* const* d_in, const int* in_sizes, int n_in,
                              void* d_out, int out_size, void* d_ws, size_t ws_size,
                              hipStream_t stream) {
  const float* x     = (const float*)d_in[0];
  const float* Wqkv  = (const float*)d_in[1];
  const float* bqkv  = (const float*)d_in[2];
  const float* Wproj = (const float*)d_in[3];
  const float* bproj = (const float*)d_in[4];
  const int*   cmask = (const int*)d_in[5];

  unsigned short* Xb     = (unsigned short*)d_ws;
  unsigned short* Wqkvb  = Xb     + (size_t)MTOT * DIM;       //  8 MB in
  unsigned short* Wprojb = Wqkvb  + (size_t)DIM * QKDIM;      // +6 MB
  unsigned short* QKVb   = Wprojb + (size_t)DIM * DIM;        // +2 MB
  unsigned short* Ob     = QKVb   + (size_t)MTOT * QKDIM;     // +24 MB
  (void)ws_size; (void)in_sizes; (void)n_in; (void)out_size;

  int nx = MTOT * DIM, nw1 = DIM * QKDIM, nw2 = DIM * DIM;
  cvt_f32_bf16<<<(nx  + 255) / 256, 256, 0, stream>>>(x,     Xb,     nx);
  cvt_f32_bf16<<<(nw1 + 255) / 256, 256, 0, stream>>>(Wqkv,  Wqkvb,  nw1);
  cvt_f32_bf16<<<(nw2 + 255) / 256, 256, 0, stream>>>(Wproj, Wprojb, nw2);

  // qkv = x @ W_qkv + b_qkv  -> bf16
  gemm_bf16<0><<<dim3(QKDIM / 64, MTOT / 128), 256, 0, stream>>>(
      Xb, Wqkvb, bqkv, QKVb, nullptr, MTOT, DIM, QKDIM);

  // flash attention -> bf16 [B*N, D]
  flash_attn<<<dim3(BD * NHEAD, SEQ / 128), 256, 0, stream>>>(QKVb, Ob, cmask);

  // out = attn_out @ W_proj + b_proj -> f32
  gemm_bf16<1><<<dim3(DIM / 64, MTOT / 128), 256, 0, stream>>>(
      Ob, Wprojb, bproj, nullptr, (float*)d_out, MTOT, DIM, DIM);
}
